// CausalIVPAttention_6390911336880
// MI455X (gfx1250) — matrix operand order repaired and checked by reading, not compiled, not run on hardware
//
#include <hip/hip_runtime.h>
#include <hip/hip_bf16.h>
#include <math.h>

// Problem constants (B=1)
#define T   512
#define Ch  256
#define H   8
#define HD  32
#define HID 64

typedef __attribute__((ext_vector_type(16))) __bf16 v16bf;
typedef __attribute__((ext_vector_type(8)))  float  v8f;
typedef __attribute__((ext_vector_type(4)))  unsigned int u32x4;
typedef __attribute__((ext_vector_type(8)))  unsigned int u32x8;

// ---- WMMA fragment loaders (contiguous k = half*16 + e in both A and B, wave32) ----

// A-matrix 16x32 bf16, row-major source with `stride` elements per row.
// lane L: m = L%16, half = L/16; element e -> k = half*16 + e -> two 16B loads.
__device__ __forceinline__ v16bf load_a_bf16(const __bf16* p0, int stride, int lane) {
  int m = lane & 15, half = lane >> 4;
  const __bf16* p = p0 + m * stride + half * 16;
  v16bf a;
  ((int4*)&a)[0] = *(const int4*)(const void*)p;
  ((int4*)&a)[1] = *(const int4*)(const void*)(p + 8);
  return a;
}

// B-matrix 32x16 bf16, memory holds B^T rows (n-major, k contiguous).
// lane L: n = L%16, half = L/16; element e -> k = half*16 + e (16 contiguous bf16).
__device__ __forceinline__ v16bf load_b_bf16(const __bf16* p0, int stride, int lane) {
  int n = lane & 15, half = lane >> 4;
  const __bf16* p = p0 + n * stride + half * 16;
  v16bf b;
  ((int4*)&b)[0] = *(const int4*)(const void*)p;
  ((int4*)&b)[1] = *(const int4*)(const void*)(p + 8);
  return b;
}

// Same B layout gathering from f32 weight matrix W[n][k]; preloaded once per wave.
__device__ __forceinline__ v16bf load_b_f32(const float* wmat, int rowlen, int ntile, int kbase, int lane) {
  int n  = ntile * 16 + (lane & 15);
  int k0 = kbase + (lane >> 4) * 16;
  const float* p = wmat + n * rowlen + k0;
  v16bf b;
#pragma unroll
  for (int e = 0; e < 16; ++e) b[e] = (__bf16)p[e];
  return b;
}

// ---- Tensor Data Mover: 2D f32 tile -> LDS (ISA 08_async_tensor §8.3-8.4) ----
// Inline asm (portable across builtin-arity differences). tile_d1 lines of tile_d0
// f32 elements, line y at global_addr + 4*y*dim0_stride; packed contiguous in LDS.
__device__ __forceinline__ void tdm_load_2d_f32(const float* gsrc, unsigned lds_off,
                                                unsigned tile_d0, unsigned tile_d1,
                                                unsigned long long dim0_stride) {
  unsigned long long ga = (unsigned long long)(size_t)gsrc;
  u32x4 g0;
  g0[0] = 1u;                                                  // count=1, user D#
  g0[1] = lds_off;                                             // lds_addr (bytes)
  g0[2] = (unsigned)ga;                                        // global_addr[31:0]
  g0[3] = (unsigned)((ga >> 32) & 0x01FFFFFFu) | (2u << 30);   // addr[56:32] | type=2
  u32x8 g1;
  g1[0] = (2u << 16);                                          // data_size = 4B
  g1[1] = (tile_d0 & 0xFFFFu) << 16;                           // tensor_dim0[15:0]
  g1[2] = ((tile_d0 >> 16) & 0xFFFFu) | ((tile_d1 & 0xFFFFu) << 16); // dim0 hi | tensor_dim1 lo
  g1[3] = ((tile_d1 >> 16) & 0xFFFFu) | ((tile_d0 & 0xFFFFu) << 16); // dim1 hi | tile_dim0
  g1[4] = (tile_d1 & 0xFFFFu);                                 // tile_dim1 (tile_dim2=0)
  g1[5] = (unsigned)dim0_stride;                               // tensor_dim0_stride lo
  g1[6] = (unsigned)((dim0_stride >> 32) & 0xFFFFu);           // stride hi (dim1_stride=0)
  g1[7] = 0u;
  u32x4 gz = {0u, 0u, 0u, 0u};
  asm volatile("tensor_load_to_lds %0, %1, %2, %3"
               :: "s"(g0), "s"(g1), "s"(gz), "s"(gz) : "memory");
}

// ---- Kernel 1: QKV projection + head split (tiny, f32) ----
__global__ void qkv_kernel(const float* __restrict__ x, const float* __restrict__ w_attn,
                           const float* __restrict__ b_attn,
                           __bf16* __restrict__ qbf, __bf16* __restrict__ kbf,
                           float* __restrict__ vf) {
  int idx = blockIdx.x * 256 + threadIdx.x;   // T * 3C threads
  int tt = idx / (3 * Ch);
  int c  = idx % (3 * Ch);
  const float* xr = x + tt * Ch;
  const float* wr = w_attn + c * Ch;
  float acc = b_attn[c];
#pragma unroll 4
  for (int d = 0; d < Ch; d += 4) {
    float4 xv = *(const float4*)(xr + d);
    float4 wv = *(const float4*)(wr + d);
    acc += xv.x * wv.x + xv.y * wv.y + xv.z * wv.z + xv.w * wv.w;
  }
  int sec = c >> 8;          // 0:q 1:k 2:v
  int cc  = c & 255;
  int h = cc >> 5, d = cc & 31;
  int o = (h * T + tt) * HD + d;
  if (sec == 0)      qbf[o] = (__bf16)acc;
  else if (sec == 1) kbf[o] = (__bf16)acc;
  else               vf[o]  = acc;
}

// ---- Kernel 2: pv[h][j][u] = W1[:, :32] @ v_j + b1 (j-only part of layer 1) ----
__global__ void pv_kernel(const float* __restrict__ vf, const float* __restrict__ w1,
                          const float* __restrict__ b1, float* __restrict__ pv) {
  int idx = blockIdx.x * 256 + threadIdx.x;   // H*T*HID
  int ht = idx / HID, u = idx % HID;
  const float* vr = vf + ht * HD;
  const float* wr = w1 + u * (HD + 1);
  float acc = b1[u];
#pragma unroll
  for (int d = 0; d < HD; ++d) acc += wr[d] * vr[d];
  pv[idx] = acc;
}

// ---- Kernel 3: attention scores (WMMA, K=HD=32 in one shot) + causal softmax ----
__global__ void attn_kernel(const __bf16* __restrict__ qbf, const __bf16* __restrict__ kbf,
                            float* __restrict__ att) {
  __shared__ float S[16 * (T + 8)];
  int tid = threadIdx.x, lane = tid & 31, w = tid >> 5;
  int half = lane >> 4, nn = lane & 15;
  int itile = blockIdx.x, h = blockIdx.y;
  int i0 = itile * 16;
  const float scale = 0.17677669529663687f;   // 1/sqrt(32)

  v16bf aq = load_a_bf16(qbf + (size_t)(h * T + i0) * HD, HD, lane);

  for (int jt = w; jt < T / 16; jt += 8) {
    v16bf bk = load_b_bf16(kbf + (size_t)(h * T + jt * 16) * HD, HD, lane);
    v8f c = {};
    c = __builtin_amdgcn_wmma_f32_16x16x32_bf16(false, aq, false, bk, (short)0, c, false, false);
    int j = jt * 16 + nn;
#pragma unroll
    for (int r = 0; r < 8; ++r) {
      int il = r + half * 8;
      float s = c[r] * scale;
      S[il * (T + 8) + j] = (j <= i0 + il) ? s : -1e30f;   // causal mask
    }
  }
  __syncthreads();

  for (int il = w; il < 16; il += 8) {
    int gi = i0 + il;
    float m = -1e30f;
    for (int j = lane; j < T; j += 32) m = fmaxf(m, S[il * (T + 8) + j]);
#pragma unroll
    for (int o = 16; o > 0; o >>= 1) m = fmaxf(m, __shfl_xor(m, o, 32));
    float sum = 0.f;
    for (int j = lane; j < T; j += 32) {
      float e = __expf(S[il * (T + 8) + j] - m);
      S[il * (T + 8) + j] = e;
      sum += e;
    }
#pragma unroll
    for (int o = 16; o > 0; o >>= 1) sum += __shfl_xor(sum, o, 32);
    float inv = 1.f / sum;
    float* arow = att + (size_t)(h * T + gi) * T;
    for (int j = lane; j < T; j += 32)
      arow[j] = (j <= gi) ? S[il * (T + 8) + j] * inv : 0.f;   // exact zeros past diagonal
  }
}

// ---- Kernel 4 (hot): pairwise ResNetFlow MLP via WMMA + weighted reduction ----
// One WG per query i; heads looped inside so the head-independent tanh time-gate
// is computed once per (i, j-chunk). Attention rows for all 8 heads staged by TDM.
__global__ void flow_kernel(const float* __restrict__ t, const float* __restrict__ pv,
                            const float* __restrict__ w1,
                            const float* __restrict__ w2, const float* __restrict__ b2,
                            const float* __restrict__ w3, const float* __restrict__ b3,
                            const float* __restrict__ tw, const float* __restrict__ tb,
                            const float* __restrict__ vf, const float* __restrict__ att,
                            float* __restrict__ y) {
  __shared__ __align__(16) float  attS[H * T];    // 16 KB, filled by tensor_load_to_lds
  __shared__ __align__(16) __bf16 H1[64 * 72];    // stride 72 -> 144B rows, 16B-aligned frags
  __shared__ __align__(16) __bf16 H2[64 * 72];
  __shared__ float RESs[64 * 33];
  __shared__ float tfS[64 * 33];                  // tanh gate, shared across all 8 heads
  __shared__ float red2[H * 8 * HD];              // per-(h, jg, d) partial sums
  __shared__ float dts[64];
  __shared__ float w1dtS[HID];
  __shared__ float twS[HD], tbS[HD];

  int tid = threadIdx.x, lane = tid & 31, w = tid >> 5;
  int half = lane >> 4, nn = lane & 15;
  int i = blockIdx.x;
  float ti = t[i];

  // TDM: stage att[h][i][:] for h=0..7 (lines of T f32, stride T*T between heads).
  if (w == 0) {
    tdm_load_2d_f32(att + (size_t)i * T, (unsigned)(size_t)&attS[0],
                    (unsigned)T, (unsigned)H, (unsigned long long)T * T);
  }

  if (tid < HID) w1dtS[tid] = w1[tid * (HD + 1) + HD];   // dt column of W1
  if (tid < HD)  { twS[tid] = tw[tid]; tbS[tid] = tb[tid]; }
  for (int idx = tid; idx < H * 8 * HD; idx += 256) red2[idx] = 0.f;

  // Per-wave fixed output tiles: GEMM2 covers 4x4 16x16 tiles, GEMM3 covers 4x2.
  int n2 = w & 3, mb2 = w >> 2;     // wave does mtiles {mb2, mb2+2}, ntile n2
  int n3 = w & 1, m3 = w >> 1;
  v16bf bW2_0 = load_b_f32(w2, HID, n2, 0,  lane);
  v16bf bW2_1 = load_b_f32(w2, HID, n2, 32, lane);
  v16bf bW3_0 = load_b_f32(w3, HID, n3, 0,  lane);
  v16bf bW3_1 = load_b_f32(w3, HID, n3, 32, lane);
  float b2l = b2[n2 * 16 + nn];
  float b3l = b3[n3 * 16 + nn];

  if (w == 0) __builtin_amdgcn_s_wait_tensorcnt(0);   // attS resident
  __syncthreads();

  int d = tid & 31, jg = tid >> 5;
  int nchunk = (i >> 6) + 1;

  for (int ch = 0; ch < nchunk; ++ch) {
    int j0 = ch * 64;
    // dt and head-shared time gate tf = tanh(dt*tw + tb), fast form via exp+rcp
    if (tid < 64) dts[tid] = ti - t[j0 + tid];
    __syncthreads();
    for (int idx = tid; idx < 64 * HD; idx += 256) {
      int jl = idx >> 5, dd = idx & 31;
      float xx = dts[jl] * twS[dd] + tbS[dd];
      float e = __expf(2.f * xx);
      tfS[jl * 33 + dd] = (e - 1.f) * __builtin_amdgcn_rcpf(e + 1.f);
    }
    __syncthreads();

    for (int hh = 0; hh < H; ++hh) {
      const float* pvh = pv + (size_t)(hh * T) * HID;
      const float* vfh = vf + (size_t)(hh * T) * HD;
      __builtin_prefetch(pvh + (size_t)j0 * HID, 0, 1);

      // H1[jl][u] = relu(pv[j][u] + dt * w1_dt[u])   (b1 folded into pv)
      for (int idx = tid; idx < 64 * HID; idx += 256) {
        int jl = idx >> 6, u = idx & 63;
        float v = pvh[(j0 + jl) * HID + u] + dts[jl] * w1dtS[u];
        H1[jl * 72 + u] = (__bf16)fmaxf(v, 0.f);
      }
      __syncthreads();

      // GEMM2: H2 = relu(H1 @ W2^T + b2), K=64 -> two 16x16x32 steps
#pragma unroll
      for (int mm = 0; mm < 2; ++mm) {
        int mt = mb2 + mm * 2;
        v16bf a0 = load_a_bf16(H1 + mt * 16 * 72, 72, lane);
        v16bf a1 = load_a_bf16(H1 + mt * 16 * 72 + 32, 72, lane);
        v8f c = {};
        c = __builtin_amdgcn_wmma_f32_16x16x32_bf16(false, a0, false, bW2_0, (short)0, c, false, false);
        c = __builtin_amdgcn_wmma_f32_16x16x32_bf16(false, a1, false, bW2_1, (short)0, c, false, false);
#pragma unroll
        for (int r = 0; r < 8; ++r) {
          float h2v = fmaxf(c[r] + b2l, 0.f);
          H2[(mt * 16 + r + half * 8) * 72 + n2 * 16 + nn] = (__bf16)h2v;
        }
      }
      __syncthreads();

      // GEMM3: RES = H2 @ W3^T + b3
      {
        v16bf a0 = load_a_bf16(H2 + m3 * 16 * 72, 72, lane);
        v16bf a1 = load_a_bf16(H2 + m3 * 16 * 72 + 32, 72, lane);
        v8f c = {};
        c = __builtin_amdgcn_wmma_f32_16x16x32_bf16(false, a0, false, bW3_0, (short)0, c, false, false);
        c = __builtin_amdgcn_wmma_f32_16x16x32_bf16(false, a1, false, bW3_1, (short)0, c, false, false);
#pragma unroll
        for (int r = 0; r < 8; ++r)
          RESs[(m3 * 16 + r + half * 8) * 33 + n3 * 16 + nn] = c[r] + b3l;
      }
      __syncthreads();

      // y[h,i,:] += att[h,i,j] * (v_j + tf * res)   (att==0 past diagonal)
      {
        float local = 0.f;
        const float* arow = &attS[hh * T];
        for (int jl = jg; jl < 64; jl += 8) {
          int j = j0 + jl;
          float aw = arow[j];
          local += aw * (vfh[j * HD + d] + tfS[jl * 33 + d] * RESs[jl * 33 + d]);
        }
        red2[(hh * 8 + jg) * HD + d] += local;
      }
      __syncthreads();
    }
  }

  // Reduce partials over jg: thread tid -> (h = tid>>5, d = tid&31)
  {
    int hh = tid >> 5, dd = tid & 31;
    float s = 0.f;
#pragma unroll
    for (int r = 0; r < 8; ++r) s += red2[(hh * 8 + r) * HD + dd];
    y[(size_t)(hh * T + i) * HD + dd] = s;
  }
}

// ---- Kernel 5: output projection (tiny, f32) ----
__global__ void proj_kernel(const float* __restrict__ y, const float* __restrict__ w_proj,
                            const float* __restrict__ b_proj, float* __restrict__ out) {
  int idx = blockIdx.x * 256 + threadIdx.x;   // T*C
  int tt = idx >> 8, c = idx & 255;
  const float* wr = w_proj + c * Ch;
  float acc = b_proj[c];
#pragma unroll 4
  for (int c2 = 0; c2 < Ch; ++c2) {
    int hh = c2 >> 5, dd = c2 & 31;
    acc += y[(size_t)(hh * T + tt) * HD + dd] * wr[c2];
  }
  out[idx] = acc;
}

extern "C" void kernel_launch(void* const* d_in, const int* in_sizes, int n_in,
                              void* d_out, int out_size, void* d_ws, size_t ws_size,
                              hipStream_t stream) {
  const float* x      = (const float*)d_in[0];
  const float* t      = (const float*)d_in[1];
  const float* w_attn = (const float*)d_in[2];
  const float* b_attn = (const float*)d_in[3];
  const float* w_proj = (const float*)d_in[4];
  const float* b_proj = (const float*)d_in[5];
  const float* w1     = (const float*)d_in[6];
  const float* b1     = (const float*)d_in[7];
  const float* w2     = (const float*)d_in[8];
  const float* b2     = (const float*)d_in[9];
  const float* w3     = (const float*)d_in[10];
  const float* b3     = (const float*)d_in[11];
  const float* tw     = (const float*)d_in[12];
  const float* tb     = (const float*)d_in[13];

  // Workspace layout (~11 MB total)
  char* ws = (char*)d_ws;
  __bf16* qbf = (__bf16*)(ws + 0);          // H*T*HD bf16   = 256 KB
  __bf16* kbf = (__bf16*)(ws + 262144);     // 256 KB
  float*  vf  = (float*) (ws + 524288);     // H*T*HD f32    = 512 KB
  float*  pv  = (float*) (ws + 1048576);    // H*T*HID f32   = 1 MB
  float*  att = (float*) (ws + 2097152);    // H*T*T f32     = 8 MB
  float*  y   = (float*) (ws + 10485760);   // H*T*HD f32    = 512 KB
  float*  out = (float*)d_out;

  qkv_kernel <<<(T * 3 * Ch) / 256, 256, 0, stream>>>(x, w_attn, b_attn, qbf, kbf, vf);
  pv_kernel  <<<(H * T * HID) / 256, 256, 0, stream>>>(vf, w1, b1, pv);
  attn_kernel<<<dim3(T / 16, H), 256, 0, stream>>>(qbf, kbf, att);
  flow_kernel<<<T, 256, 0, stream>>>(t, pv, w1, w2, b2, w3, b3, tw, tb, vf, att, y);
  proj_kernel<<<(T * Ch) / 256, 256, 0, stream>>>(y, w_proj, b_proj, out);
}
